// LayerStacks_4346506903761
// MI455X (gfx1250) — compile-verified, matrix operand order checked by reading, not run on hardware
//
#include <hip/hip_runtime.h>
#include <hip/hip_bf16.h>

typedef __attribute__((ext_vector_type(16))) _Float16 v16h;
typedef __attribute__((ext_vector_type(8)))  float    v8f;

#define DIM1    3072
#define NEXP    8
#define NSAMP   4096
#define IBLK    256          // i-values per staged block
#define NBLK    12           // 3072 / 256
#define NFEAT   9            // 8 spline basis + silu
#define CHUNKS  72           // 9*256/32 K-chunks of 32 per block
#define CPW     9            // contiguous chunks per wave (72/8)
#define L2W     16           // layer-1 out width (L2+1)
#define L3W     32
#define NWAVE   8            // waves per workgroup (K-split)

// ---- fragment index helpers (V_WMMA_F32_16X16X32_F16 layouts) -------------
// A 16x32 f16: lane L(0-15) row M=L holds kk {0..7,16..23}; lane L+16 holds
// kk {8..15,24..31}; within lane, f16 slot j: kk = (j&7) + 8*(L>>4) + 16*(j>>3)
__device__ __forceinline__ int kk_of(int lane, int j) {
    return (j & 7) + ((lane >> 4) << 3) + ((j >> 3) << 4);
}
// scatter position for A value (k-feature, i-in-block, sample)
__device__ __forceinline__ int frag_idx(int k, int il, int s) {
    int c  = k * (IBLK / 32) + (il >> 5);
    int kk = il & 31;
    int ln = s + (((kk >> 3) & 1) << 4);
    int j  = (kk & 7) + ((kk >> 4) << 3);
    return (c * 32 + ln) * 16 + j;
}

// uniform cubic B-spline (cardinal) weights; matches Cox-de Boor on the
// uniform extended knot grid of the reference (GRID=5, K=3, h=0.4)
__device__ __forceinline__ void cubw(float x, int& mm, float w[4]) {
    float v  = (x + 1.0f) * 2.5f + 3.0f;   // (x - t0)/h
    float fm = floorf(v);
    mm = (int)fm;
    float t = v - fm, t2 = t * t, t3 = t2 * t, o = 1.0f - t;
    w[0] = t3 * (1.0f / 6.0f);                                        // kidx = mm
    w[1] = (1.0f + 3.0f * t + 3.0f * t2 - 3.0f * t3) * (1.0f / 6.0f); // mm-1
    w[2] = (4.0f - 6.0f * t2 + 3.0f * t3) * (1.0f / 6.0f);            // mm-2
    w[3] = o * o * o * (1.0f / 6.0f);                                 // mm-3
}
__device__ __forceinline__ float basis1(float x, int k) {
    int mm; float w[4]; cubw(x, mm, w);
    int d = mm - k;
    return (d >= 0 && d < 4) ? w[d] : 0.0f;
}
__device__ __forceinline__ float silu(float x) { return x / (1.0f + __expf(-x)); }

// ---------------------------------------------------------------------------
__global__ void k_init(int* counts) { if (threadIdx.x < NEXP) counts[threadIdx.x] = 0; }

__global__ void k_route(const int* __restrict__ ls, int* counts, int* order) {
    int n = blockIdx.x * 256 + threadIdx.x;
    if (n >= NSAMP) return;
    int e = ls[n];
    int p = atomicAdd(&counts[e], 1);
    order[e * NSAMP + p] = n;
}

// Fuse layer-1 weights (coef*ws, wb as feature 8) into B-fragment lane order:
// flat = (((e*NBLK + ib)*CHUNKS + c)*32 + lane)*16 + j
__global__ void k_fuse1(const float* __restrict__ coef1, const float* __restrict__ wb1,
                        const float* __restrict__ ws1, _Float16* __restrict__ W1f) {
    int idx  = blockIdx.x * 256 + threadIdx.x;        // 8*12*72*512 = 3,538,944
    int j    = idx & 15;
    int lane = (idx >> 4) & 31;
    int rest = idx >> 9;
    int c    = rest % CHUNKS; rest /= CHUNKS;
    int ib   = rest % NBLK;
    int e    = rest / NBLK;
    int kk = kk_of(lane, j);
    int k  = c >> 3;                         // feature (0..8)
    int i  = ib * IBLK + ((c & 7) << 5) + kk;
    int o  = lane & 15;
    int base = (e * DIM1 + i) * L2W + o;
    float v = (k < 8) ? coef1[base * 8 + k] * ws1[base] : wb1[base];
    W1f[idx] = (_Float16)v;
}

// Layer-2 fused weights, padded K 144->160, both N-halves:
// flat = (((e*2 + half)*5 + c2)*32 + lane)*16 + j
__global__ void k_fuse2(const float* __restrict__ coef2, const float* __restrict__ wb2,
                        const float* __restrict__ ws2, _Float16* __restrict__ W2f) {
    int idx  = blockIdx.x * 256 + threadIdx.x;        // 8*2*5*512 = 40,960
    int j    = idx & 15;
    int lane = (idx >> 4) & 31;
    int rest = idx >> 9;
    int c2   = rest % 5; rest /= 5;
    int half = rest & 1;
    int e    = rest >> 1;
    int f = c2 * 32 + kk_of(lane, j);
    float v = 0.0f;
    if (f < NFEAT * L2W) {
        int k = f >> 4, i = f & 15;
        int o = half * 16 + (lane & 15);
        int base = (e * L2W + i) * L3W + o;
        v = (k < 8) ? coef2[base * 8 + k] * ws2[base] : wb2[base];
    }
    W2f[idx] = (_Float16)v;
}

// Fused: basis-gen -> layer1 WMMA (K=27648, 9 contiguous chunks per wave)
// -> layer2 WMMA -> layer3 scalar -> routed scatter
__global__ __launch_bounds__(256) void k_main(
    const float* __restrict__ x,
    const float* __restrict__ coef3, const float* __restrict__ wb3,
    const float* __restrict__ ws3,
    const _Float16* __restrict__ W1f, const _Float16* __restrict__ W2f,
    const int* __restrict__ counts, const int* __restrict__ order,
    float* __restrict__ out) {
    __shared__ _Float16 afrag[CHUNKS * 32 * 16] __attribute__((aligned(32)));  // 72 KB
    __shared__ float cpart[NWAVE][16][16];
    __shared__ float h1t[16][16];
    __shared__ float h2t[16][32];
    __shared__ int   sidx[16];

    const int e = blockIdx.x >> 8;
    const int tile = blockIdx.x & 255;
    const int cnt = counts[e];
    if (tile * 16 >= cnt) return;

    const int tid  = threadIdx.x;
    const int lane = tid & 31;
    // scalarize wave index: chunk range becomes SGPR-uniform (pure s_cbranch),
    // EXEC provably all-1s at every WMMA (ISA requirement)
    const int wv = __builtin_amdgcn_readfirstlane(tid >> 5);

    if (tid < 16) {
        int g = tile * 16 + tid;
        sidx[tid] = order[e * NSAMP + (g < cnt ? g : cnt - 1)];
    }
    __syncthreads();

    v8f acc = {};
    const int il = tid;                // i within block (0..255)
    for (int ib = 0; ib < NBLK; ++ib) {
        // zero A-fragment staging (72 KB)
        uint4 zz = {0u, 0u, 0u, 0u};
        uint4* az = (uint4*)afrag;
        #pragma unroll
        for (int z = 0; z < 18; ++z) az[z * 256 + tid] = zz;
        __syncthreads();
        // generate basis+silu for 256 i x 16 samples, scatter in frag order
        for (int s = 0; s < 16; ++s) {
            float xv = x[(size_t)sidx[s] * DIM1 + ib * IBLK + il];
            int mm; float w[4]; cubw(xv, mm, w);
            #pragma unroll
            for (int d = 0; d < 4; ++d) {
                int kidx = mm - d;
                if (kidx >= 0 && kidx < 8)
                    afrag[frag_idx(kidx, il, s)] = (_Float16)w[d];
            }
            afrag[frag_idx(8, il, s)] = (_Float16)silu(xv);
        }
        __syncthreads();
        // 9 contiguous K-chunks per wave: A from LDS, B coalesced from global
        const _Float16* wbase = W1f + (size_t)((e * NBLK + ib) * CHUNKS) * 512;
        __builtin_prefetch(wbase + CHUNKS * 512, 0, 1);
        #pragma unroll 3
        for (int c = wv * CPW; c < wv * CPW + CPW; ++c) {
            v16h a = *(const v16h*)(afrag + c * 512 + lane * 16);
            v16h b = *(const v16h*)(wbase + c * 512 + lane * 16);
            acc = __builtin_amdgcn_wmma_f32_16x16x32_f16(
                false, a, false, b, (short)0, acc, false, false);
        }
        __syncthreads();
    }
    // reduce K-split partials across the 8 waves
    {
        int r0 = (lane >> 4) * 8, col = lane & 15;
        #pragma unroll
        for (int r = 0; r < 8; ++r) cpart[wv][r0 + r][col] = acc[r];
    }
    __syncthreads();

    if (wv == 0) {
        for (int idx = lane; idx < 256; idx += 32) {
            int rr = idx >> 4, cc = idx & 15;
            float s = 0.0f;
            #pragma unroll
            for (int w = 0; w < NWAVE; ++w) s += cpart[w][rr][cc];
            h1t[rr][cc] = s;
        }
        // ---- layer 2: K=144 (pad 160), out 32 = two WMMA C tiles ----
        v8f a20 = {}, a21 = {};
        const int m = lane & 15;
        for (int c2 = 0; c2 < 5; ++c2) {
            v16h af;
            #pragma unroll
            for (int j = 0; j < 16; ++j) {
                int f = c2 * 32 + kk_of(lane, j);
                float v = 0.0f;
                if (f < NFEAT * L2W) {
                    int k = f >> 4, i = f & 15;
                    float h = h1t[m][i];
                    v = (k == 8) ? silu(h) : basis1(h, k);
                }
                af[j] = (_Float16)v;
            }
            const v16h* b0 = (const v16h*)(W2f + (size_t)(((e * 2 + 0) * 5 + c2) * 32 + lane) * 16);
            const v16h* b1 = (const v16h*)(W2f + (size_t)(((e * 2 + 1) * 5 + c2) * 32 + lane) * 16);
            a20 = __builtin_amdgcn_wmma_f32_16x16x32_f16(false, af, false, *b0, (short)0, a20, false, false);
            a21 = __builtin_amdgcn_wmma_f32_16x16x32_f16(false, af, false, *b1, (short)0, a21, false, false);
        }
        {
            int r0 = (lane >> 4) * 8, col = lane & 15;
            #pragma unroll
            for (int r = 0; r < 8; ++r) {
                h2t[r0 + r][col]      = a20[r];
                h2t[r0 + r][16 + col] = a21[r];
            }
        }
        // ---- layer 3: 32 -> 1, scalar per sample-lane ----
        if (lane < 16) {
            float acc3 = 0.0f;
            for (int i = 0; i < L3W; ++i) {
                float h = h2t[lane][i];
                int mm; float w[4]; cubw(h, mm, w);
                const float* c3 = coef3 + (e * L3W + i) * 8;
                float sp = 0.0f;
                #pragma unroll
                for (int d = 0; d < 4; ++d) {
                    int kidx = mm - d;
                    if (kidx >= 0 && kidx < 8) sp += w[d] * c3[kidx];
                }
                acc3 += ws3[e * L3W + i] * sp + wb3[e * L3W + i] * silu(h);
            }
            if (tile * 16 + lane < cnt) out[sidx[lane]] = acc3;
        }
    }
}

extern "C" void kernel_launch(void* const* d_in, const int* in_sizes, int n_in,
                              void* d_out, int out_size, void* d_ws, size_t ws_size,
                              hipStream_t stream) {
    const float* x     = (const float*)d_in[0];
    const int*   ls    = (const int*)d_in[1];
    const float* coef1 = (const float*)d_in[2];
    const float* wb1   = (const float*)d_in[3];
    const float* ws1   = (const float*)d_in[4];
    const float* coef2 = (const float*)d_in[5];
    const float* wb2   = (const float*)d_in[6];
    const float* ws2   = (const float*)d_in[7];
    const float* coef3 = (const float*)d_in[8];
    const float* wb3   = (const float*)d_in[9];
    const float* ws3   = (const float*)d_in[10];
    float* out = (float*)d_out;

    // workspace carve: counts(32B) | order(128KB) | W1f(6.75MB f16) | W2f(80KB f16)
    char* ws = (char*)d_ws;
    int* counts   = (int*)ws;
    int* order    = (int*)(ws + 32);
    _Float16* W1f = (_Float16*)(ws + 131136);
    _Float16* W2f = (_Float16*)(ws + 131136 + 7077888);

    k_init <<<1, 32, 0, stream>>>(counts);
    k_route<<<NSAMP / 256, 256, 0, stream>>>(ls, counts, order);
    k_fuse1<<<(NEXP * NBLK * CHUNKS * 512) / 256, 256, 0, stream>>>(coef1, wb1, ws1, W1f);
    k_fuse2<<<(NEXP * 2 * 5 * 512) / 256, 256, 0, stream>>>(coef2, wb2, ws2, W2f);
    k_main <<<NEXP * 256, 256, 0, stream>>>(x, coef3, wb3, ws3, W1f, W2f, counts, order, out);
}
